// MoELayer_62818191671881
// MI455X (gfx1250) — matrix operand order repaired and checked by reading, not compiled, not run on hardware
//
#include <hip/hip_runtime.h>

// Problem constants (match reference)
#define N_TOK 8192
#define DIMD  2048
#define HDIM  8192
#define ODIM  2048
#define NEXP  8

typedef __attribute__((ext_vector_type(8)))  float          v8f;
typedef __attribute__((ext_vector_type(16))) __bf16         v16bf;
typedef __attribute__((ext_vector_type(2)))  __bf16         v2bf;
typedef __attribute__((ext_vector_type(8)))  unsigned short us8;
typedef __attribute__((ext_vector_type(16))) unsigned short us16;

// fp32 -> bf16 bits, round-to-nearest-even (scalar fallback / epilogue)
__device__ __forceinline__ unsigned short f2bf(float f) {
  unsigned u = __builtin_bit_cast(unsigned, f);
  u += 0x7FFFu + ((u >> 16) & 1u);
  return (unsigned short)(u >> 16);
}

// Pack two fp32 into two bf16 (one dword). Native v_cvt_pk_bf16_f32 if available.
__device__ __forceinline__ unsigned pack2bf(float lo, float hi) {
#if __has_builtin(__builtin_amdgcn_cvt_pk_bf16_f32)
  v2bf r = __builtin_amdgcn_cvt_pk_bf16_f32(lo, hi);
  return __builtin_bit_cast(unsigned, r);
#else
  return (unsigned)f2bf(lo) | ((unsigned)f2bf(hi) << 16);
#endif
}

__device__ __forceinline__ v16bf make_frag(us8 lo, us8 hi) {
  us16 v = __builtin_shufflevector(lo, hi, 0, 1, 2, 3, 4, 5, 6, 7,
                                   8, 9, 10, 11, 12, 13, 14, 15);
  return __builtin_bit_cast(v16bf, v);
}

// CDNA5 async copy: 16B global -> LDS, tracked by ASYNCcnt (per-lane, GV mode).
// LDS address = low 32 bits of the generic pointer (ISA aperture rule).
__device__ __forceinline__ void async_copy_b128(void* lds, const void* gptr) {
  unsigned l = (unsigned)(unsigned long long)(uintptr_t)lds;
  unsigned long long g = (unsigned long long)(uintptr_t)gptr;
  asm volatile("global_load_async_to_lds_b128 %0, %1, off"
               :
               : "v"(l), "v"(g)
               : "memory");
}
__device__ __forceinline__ void wait_async0() {
  asm volatile("s_wait_asynccnt 0" ::: "memory");
}

// ---------------- init / zero ----------------
__global__ void k_init(float* P, int* counts) {
  int t = threadIdx.x;
  if (t < NEXP) { P[t] = 0.f; counts[t] = 0; }
}

__global__ void k_zero(float4* out, int n4) {
  int i = blockIdx.x * blockDim.x + threadIdx.x;
  if (i < n4) out[i] = make_float4(0.f, 0.f, 0.f, 0.f);
}

// ---------------- router: logits -> softmax -> top2 -> renorm ----------------
__global__ __launch_bounds__(256) void k_router(
    const float* __restrict__ x, const float* __restrict__ gW,
    const float* __restrict__ gb, float* __restrict__ P,
    int2* __restrict__ tki, float2* __restrict__ tkw) {
  int wid = threadIdx.x >> 5, lane = threadIdx.x & 31;
  int t = blockIdx.x * 8 + wid;
  float a[NEXP];
#pragma unroll
  for (int e = 0; e < NEXP; ++e) a[e] = 0.f;
  const float* xr = x + (size_t)t * DIMD;
  for (int d = lane; d < DIMD; d += 32) {
    float xv = xr[d];
    const float4* g = reinterpret_cast<const float4*>(gW + (size_t)d * NEXP);
    float4 g0 = g[0], g1 = g[1];
    a[0] += xv * g0.x; a[1] += xv * g0.y; a[2] += xv * g0.z; a[3] += xv * g0.w;
    a[4] += xv * g1.x; a[5] += xv * g1.y; a[6] += xv * g1.z; a[7] += xv * g1.w;
  }
#pragma unroll
  for (int e = 0; e < NEXP; ++e) {
#pragma unroll
    for (int off = 16; off > 0; off >>= 1) a[e] += __shfl_down(a[e], off);
  }
  if (lane == 0) {
    float mx = -3.4e38f;
#pragma unroll
    for (int e = 0; e < NEXP; ++e) { a[e] += gb[e]; mx = fmaxf(mx, a[e]); }
    float s = 0.f;
#pragma unroll
    for (int e = 0; e < NEXP; ++e) { a[e] = __expf(a[e] - mx); s += a[e]; }
    float inv = 1.f / s;
    int i0 = 0, i1 = 0; float p0 = -1.f, p1 = -1.f;
#pragma unroll
    for (int e = 0; e < NEXP; ++e) {
      float p = a[e] * inv;
      atomicAdd(&P[e], p);
      if (p > p0) { p1 = p0; i1 = i0; p0 = p; i0 = e; }
      else if (p > p1) { p1 = p; i1 = e; }
    }
    float rs = 1.f / (p0 + p1);
    tki[t] = make_int2(i0, i1);
    tkw[t] = make_float2(p0 * rs, p1 * rs);
  }
}

// ---------------- dispatch: build per-expert token lists ----------------
__global__ void k_dispatch(const int2* __restrict__ tki,
                           const float2* __restrict__ tkw,
                           int* __restrict__ counts, int* __restrict__ atok,
                           float* __restrict__ aw) {
  int t = blockIdx.x * blockDim.x + threadIdx.x;
  if (t >= N_TOK) return;
  int2 ii = tki[t];
  float2 ww = tkw[t];
  int s0 = atomicAdd(&counts[ii.x], 1);
  atok[ii.x * N_TOK + s0] = t;
  aw[ii.x * N_TOK + s0] = ww.x;
  int s1 = atomicAdd(&counts[ii.y], 1);
  atok[ii.y * N_TOK + s1] = t;
  aw[ii.y * N_TOK + s1] = ww.y;
}

// ---------------- prefix offsets + lb_loss ----------------
__global__ void k_finalize(const int* __restrict__ counts, int* __restrict__ offs,
                           const float* __restrict__ P, float* __restrict__ lb_out) {
  if (threadIdx.x == 0) {
    int acc = 0;
    for (int e = 0; e < NEXP; ++e) { offs[e] = acc; acc += counts[e]; }
    float lb = 0.f;
    for (int e = 0; e < NEXP; ++e) {
      float p = P[e] / (float)N_TOK;
      lb += p * p;
    }
    *lb_out = lb * (float)NEXP;
  }
}

// ---------------- grouped GEMM 1: h = relu(Xe @ fc1^T + b) -> bf16 ----------------
// 256 thr = 8 waves; C-tile 128x128; K-step 32; double-buffered LDS.
__global__ __launch_bounds__(256) void k_fc1(
    const float* __restrict__ x, const float* __restrict__ W,
    const float* __restrict__ b, const int* __restrict__ counts,
    const int* __restrict__ offs, const int* __restrict__ atok,
    unsigned short* __restrict__ hbuf) {
  int e = blockIdx.z;
  int cnt = counts[e];
  int m0 = blockIdx.x * 128;
  if (m0 >= cnt) return;
  int h0 = blockIdx.y * 128;

  __shared__ alignas(64) unsigned short As[2][128][32];
  __shared__ alignas(64) unsigned short Bs[2][128][32];
  __shared__ int toks[128];

  int tid = threadIdx.x;
  if (tid < 128) {
    int s = m0 + tid;
    toks[tid] = atok[e * N_TOK + (s < cnt ? s : cnt - 1)];
  }
  __syncthreads();

  int lane = tid & 31, wid = tid >> 5;
  int wm = wid & 1, wn = wid >> 1;  // wave tile: rows [wm*64,+64), cols [wn*32,+32)
  int row = tid >> 1;               // staging row 0..127
  int part = (tid & 1) << 4;        // staging k-half: 0..15 / 16..31

  const float* axr = x + (size_t)toks[row] * DIMD + part;
  const float* bwr = W + ((size_t)e * HDIM + h0 + row) * DIMD + part;

  v8f acc[4][2];
  {
    v8f zf = {};
#pragma unroll
    for (int i = 0; i < 4; ++i)
#pragma unroll
      for (int j = 0; j < 2; ++j) acc[i][j] = zf;
  }

  auto stage = [&](int buf, int k0) {
    __builtin_prefetch(axr + k0 + 32, 0, 1);
    __builtin_prefetch(bwr + k0 + 32, 0, 1);
    const float4* a4 = reinterpret_cast<const float4*>(axr + k0);
    float4 f0 = a4[0], f1 = a4[1], f2 = a4[2], f3 = a4[3];
    uint4 lo = make_uint4(pack2bf(f0.x, f0.y), pack2bf(f0.z, f0.w),
                          pack2bf(f1.x, f1.y), pack2bf(f1.z, f1.w));
    uint4 hi = make_uint4(pack2bf(f2.x, f2.y), pack2bf(f2.z, f2.w),
                          pack2bf(f3.x, f3.y), pack2bf(f3.z, f3.w));
    *reinterpret_cast<uint4*>(&As[buf][row][part]) = lo;
    *reinterpret_cast<uint4*>(&As[buf][row][part + 8]) = hi;
    const float4* b4 = reinterpret_cast<const float4*>(bwr + k0);
    f0 = b4[0]; f1 = b4[1]; f2 = b4[2]; f3 = b4[3];
    lo = make_uint4(pack2bf(f0.x, f0.y), pack2bf(f0.z, f0.w),
                    pack2bf(f1.x, f1.y), pack2bf(f1.z, f1.w));
    hi = make_uint4(pack2bf(f2.x, f2.y), pack2bf(f2.z, f2.w),
                    pack2bf(f3.x, f3.y), pack2bf(f3.z, f3.w));
    *reinterpret_cast<uint4*>(&Bs[buf][row][part]) = lo;
    *reinterpret_cast<uint4*>(&Bs[buf][row][part + 8]) = hi;
  };

  auto compute = [&](int buf) {
    int c0 = (lane & 16) ? 16 : 0;  // A-frag K half per lane group
    int kb = (lane & 16) ? 16 : 0;  // B-frag K half per lane group
    int ml = lane & 15;
    v16bf bfr[2];
#pragma unroll
    for (int nf = 0; nf < 2; ++nf) {
      int n = wn * 32 + nf * 16 + ml;
      bfr[nf] = __builtin_bit_cast(v16bf, *(const us16*)&Bs[buf][n][kb]);
    }
#pragma unroll
    for (int mf = 0; mf < 4; ++mf) {
      int m = wm * 64 + mf * 16 + ml;
      v16bf af =
          make_frag(*(const us8*)&As[buf][m][c0], *(const us8*)&As[buf][m][c0 + 8]);
#pragma unroll
      for (int nf = 0; nf < 2; ++nf) {
        acc[mf][nf] = __builtin_amdgcn_wmma_f32_16x16x32_bf16(
            false, af, false, bfr[nf], (short)0, acc[mf][nf], false, false);
      }
    }
  };

  const int nsteps = DIMD / 32;
  stage(0, 0);
  __syncthreads();
  for (int kt = 0; kt < nsteps; ++kt) {
    int cur = kt & 1;
    if (kt + 1 < nsteps) stage(cur ^ 1, (kt + 1) * 32);  // overlap with WMMA
    compute(cur);
    __syncthreads();
  }

  // epilogue: +bias, relu, store bf16 hidden row
  int offe = offs[e];
  const float* be = b + (size_t)e * HDIM;
#pragma unroll
  for (int mf = 0; mf < 4; ++mf) {
#pragma unroll
    for (int nf = 0; nf < 2; ++nf) {
      int col = h0 + wn * 32 + nf * 16 + (lane & 15);
      float bias = be[col];
#pragma unroll
      for (int r = 0; r < 8; ++r) {
        int m = wm * 64 + mf * 16 + ((lane >> 4) << 3) + r;
        int s = m0 + m;
        if (s < cnt) {
          float v = fmaxf(acc[mf][nf][r] + bias, 0.f);
          hbuf[(size_t)(offe + s) * HDIM + col] = f2bf(v);
        }
      }
    }
  }
}

// ---------------- grouped GEMM 2: out += w * (h @ fc2^T + b) ----------------
// A operand (bf16 hidden) staged with CDNA5 async global->LDS copies.
__global__ __launch_bounds__(256) void k_fc2(
    const unsigned short* __restrict__ hbuf, const float* __restrict__ W,
    const float* __restrict__ b, const int* __restrict__ counts,
    const int* __restrict__ offs, const int* __restrict__ atok,
    const float* __restrict__ aw, float* __restrict__ out) {
  int e = blockIdx.z;
  int cnt = counts[e];
  int m0 = blockIdx.x * 128;
  if (m0 >= cnt) return;
  int o0 = blockIdx.y * 128;

  __shared__ alignas(64) unsigned short As[2][128][32];
  __shared__ alignas(64) unsigned short Bs[2][128][32];
  __shared__ int ttok[128];
  __shared__ float twt[128];

  int tid = threadIdx.x;
  if (tid < 128) {
    int s = m0 + tid;
    int sc = (s < cnt ? s : cnt - 1);
    ttok[tid] = atok[e * N_TOK + sc];
    twt[tid] = aw[e * N_TOK + sc];
  }
  __syncthreads();

  int lane = tid & 31, wid = tid >> 5;
  int wm = wid & 1, wn = wid >> 1;
  int row = tid >> 1;
  int part = (tid & 1) << 4;

  int offe = offs[e];
  int srow = m0 + row;
  srow = (srow < cnt ? srow : cnt - 1);
  const unsigned short* axr = hbuf + (size_t)(offe + srow) * HDIM + part;
  const float* bwr = W + ((size_t)e * ODIM + o0 + row) * HDIM + part;

  v8f acc[4][2];
  {
    v8f zf = {};
#pragma unroll
    for (int i = 0; i < 4; ++i)
#pragma unroll
      for (int j = 0; j < 2; ++j) acc[i][j] = zf;
  }

  auto stage = [&](int buf, int k0) {
    __builtin_prefetch(bwr + k0 + 32, 0, 1);
    // A: bf16 already -> async 16B copies straight into LDS (no VGPR round-trip)
    async_copy_b128(&As[buf][row][part], axr + k0);
    async_copy_b128(&As[buf][row][part + 8], axr + k0 + 8);
    // B: fp32 -> bf16 pack, ds_store_b128
    const float4* b4 = reinterpret_cast<const float4*>(bwr + k0);
    float4 f0 = b4[0], f1 = b4[1], f2 = b4[2], f3 = b4[3];
    uint4 lo = make_uint4(pack2bf(f0.x, f0.y), pack2bf(f0.z, f0.w),
                          pack2bf(f1.x, f1.y), pack2bf(f1.z, f1.w));
    uint4 hi = make_uint4(pack2bf(f2.x, f2.y), pack2bf(f2.z, f2.w),
                          pack2bf(f3.x, f3.y), pack2bf(f3.z, f3.w));
    *reinterpret_cast<uint4*>(&Bs[buf][row][part]) = lo;
    *reinterpret_cast<uint4*>(&Bs[buf][row][part + 8]) = hi;
  };

  auto compute = [&](int buf) {
    int c0 = (lane & 16) ? 16 : 0;
    int kb = (lane & 16) ? 16 : 0;
    int ml = lane & 15;
    v16bf bfr[2];
#pragma unroll
    for (int nf = 0; nf < 2; ++nf) {
      int n = wn * 32 + nf * 16 + ml;
      bfr[nf] = __builtin_bit_cast(v16bf, *(const us16*)&Bs[buf][n][kb]);
    }
#pragma unroll
    for (int mf = 0; mf < 4; ++mf) {
      int m = wm * 64 + mf * 16 + ml;
      v16bf af =
          make_frag(*(const us8*)&As[buf][m][c0], *(const us8*)&As[buf][m][c0 + 8]);
#pragma unroll
      for (int nf = 0; nf < 2; ++nf) {
        acc[mf][nf] = __builtin_amdgcn_wmma_f32_16x16x32_bf16(
            false, af, false, bfr[nf], (short)0, acc[mf][nf], false, false);
      }
    }
  };

  const int nsteps = HDIM / 32;
  stage(0, 0);
  wait_async0();
  __syncthreads();
  for (int kt = 0; kt < nsteps; ++kt) {
    int cur = kt & 1;
    if (kt + 1 < nsteps) stage(cur ^ 1, (kt + 1) * 32);
    compute(cur);       // async copies for next tile overlap the WMMAs
    wait_async0();      // publish next buffer's async LDS writes
    __syncthreads();
  }

  // epilogue: y = acc + bias; out[token] += w * y  (f32 global atomics)
  const float* be = b + (size_t)e * ODIM;
#pragma unroll
  for (int mf = 0; mf < 4; ++mf) {
#pragma unroll
    for (int nf = 0; nf < 2; ++nf) {
      int col = o0 + wn * 32 + nf * 16 + (lane & 15);
      float bias = be[col];
#pragma unroll
      for (int r = 0; r < 8; ++r) {
        int m = wm * 64 + mf * 16 + ((lane >> 4) << 3) + r;
        int s = m0 + m;
        if (s < cnt) {
          float y = acc[mf][nf][r] + bias;
          atomicAdd(&out[(size_t)ttok[m] * ODIM + col], twt[m] * y);
        }
      }
    }
  }
}

// ---------------- host launcher ----------------
extern "C" void kernel_launch(void* const* d_in, const int* in_sizes, int n_in,
                              void* d_out, int out_size, void* d_ws, size_t ws_size,
                              hipStream_t stream) {
  (void)in_sizes; (void)n_in; (void)out_size; (void)ws_size;
  const float* x    = (const float*)d_in[0];
  const float* gW   = (const float*)d_in[1];
  const float* gb   = (const float*)d_in[2];
  const float* fc1W = (const float*)d_in[3];
  const float* fc1b = (const float*)d_in[4];
  const float* fc2W = (const float*)d_in[5];
  const float* fc2b = (const float*)d_in[6];
  float* out = (float*)d_out;

  // Workspace carve-up (~270 MB; dominated by the bf16 hidden buffer)
  char* ws = (char*)d_ws;
  size_t off = 0;
  auto take = [&](size_t bytes) {
    char* p = ws + off;
    off += (bytes + 255) & ~(size_t)255;
    return p;
  };
  float* P      = (float*)take(NEXP * sizeof(float));
  int*   counts = (int*)take(NEXP * sizeof(int));
  int*   offs   = (int*)take(NEXP * sizeof(int));
  int2*  tki    = (int2*)take((size_t)N_TOK * sizeof(int2));
  float2* tkw   = (float2*)take((size_t)N_TOK * sizeof(float2));
  int*   atok   = (int*)take((size_t)NEXP * N_TOK * sizeof(int));
  float* aw     = (float*)take((size_t)NEXP * N_TOK * sizeof(float));
  unsigned short* hbuf =
      (unsigned short*)take((size_t)2 * N_TOK * HDIM * sizeof(unsigned short));

  hipLaunchKernelGGL(k_init, dim3(1), dim3(32), 0, stream, P, counts);
  int n4 = (N_TOK * ODIM) / 4;
  hipLaunchKernelGGL(k_zero, dim3(n4 / 256), dim3(256), 0, stream,
                     (float4*)out, n4);
  hipLaunchKernelGGL(k_router, dim3(N_TOK / 8), dim3(256), 0, stream,
                     x, gW, gb, P, tki, tkw);
  hipLaunchKernelGGL(k_dispatch, dim3(N_TOK / 256), dim3(256), 0, stream,
                     tki, tkw, counts, atok, aw);
  hipLaunchKernelGGL(k_finalize, dim3(1), dim3(32), 0, stream,
                     counts, offs, P, out + (size_t)N_TOK * ODIM);
  hipLaunchKernelGGL(k_fc1, dim3(N_TOK / 128, HDIM / 128, NEXP), dim3(256), 0,
                     stream, x, fc1W, fc1b, counts, offs, atok, hbuf);
  hipLaunchKernelGGL(k_fc2, dim3(N_TOK / 128, ODIM / 128, NEXP), dim3(256), 0,
                     stream, hbuf, fc2W, fc2b, counts, offs, atok, aw, out);
}
